// LSTM_17841294147975
// MI455X (gfx1250) — matrix operand order repaired and checked by reading, not compiled, not run on hardware
//
#include <hip/hip_runtime.h>
#include <hip/hip_bf16.h>
#include <stdint.h>

// ---------------- CDNA5 WMMA types ----------------
typedef __attribute__((ext_vector_type(16))) _Float16 v16h;
typedef __attribute__((ext_vector_type(8)))  _Float16 half8;
typedef __attribute__((ext_vector_type(8)))  float    v8f;

#define T_DIM 512
#define B_DIM 32
#define D_IN  512
#define H_DIM 512
#define G_DIM 2048   // 4*H
#define NWG   8
#define HS    64     // H / NWG

// ---- A fragment (16x32 f16, M x K) from row-major f16 src; base -> (m0,k0) elem
static __device__ inline v16h load_frag_A(const _Float16* __restrict__ base, int ld, int lane) {
  // lanes 0-15: M=lane, K = {0..7, 16..23}; lanes 16-31: M=lane-16, K = {8..15, 24..31}
  int r  = lane & 15;
  int kb = (lane >> 4) << 3;            // 0 or 8
  const _Float16* p = base + (size_t)r * ld + kb;
  union { v16h v; half8 h[2]; } u;
  u.h[0] = *(const half8*)(p);
  u.h[1] = *(const half8*)(p + 16);
  return u.v;
}

// ---- B fragment (32x16 f16, K x N) where B[k][n] = W[n0+n][k0+k], W row-major.
// Column n of B == a contiguous row of W. lanes 0-15 hold K={0..7,16..23}, lanes 16-31 K={8..15,24..31}.
static __device__ inline v16h load_frag_B(const _Float16* __restrict__ base, int ld, int lane) {
  const _Float16* p = base + (size_t)(lane & 15) * ld + ((lane >> 4) << 3);
  union { v16h v; half8 h[2]; } u;
  u.h[0] = *(const half8*)(p);
  u.h[1] = *(const half8*)(p + 16);
  return u.v;
}

static __device__ inline float sigmoidf_(float x) { return 1.0f / (1.0f + expf(-x)); }

// ---------------- utility kernels ----------------
__global__ void zero32_kernel(uint32_t* p, int n) {
  int i = blockIdx.x * blockDim.x + threadIdx.x;
  if (i < n) p[i] = 0u;
}

__global__ void f32_to_f16_kernel(const float* __restrict__ s, _Float16* __restrict__ d, int n) {
  int i = blockIdx.x * blockDim.x + threadIdx.x;
  if (i < n) d[i] = (_Float16)s[i];
}

// ---------------- input-projection GEMM ----------------
// out[m][n] = bias[n] + sum_k A[m][k] * W[n][k]     (A: MxK f16, W: NxK f16, out fp32)
// block = 8 waves; block tile 128(M) x 128(N); wave tile 32(M) x 64(N).
__global__ __launch_bounds__(256)
void gemm_xw_kernel(const _Float16* __restrict__ A, const _Float16* __restrict__ W,
                    const float* __restrict__ bias, float* __restrict__ out,
                    int M, int N, int K) {
  const int lane = threadIdx.x & 31;
  const int wave = threadIdx.x >> 5;
  const int m0 = blockIdx.y * 128 + (wave & 3) * 32;
  const int n0 = blockIdx.x * 128 + (wave >> 2) * 64;

  v8f acc[2][4];
  for (int j = 0; j < 4; ++j) {
    float bj = bias[n0 + 16 * j + (lane & 15)];
    v8f b8 = { bj, bj, bj, bj, bj, bj, bj, bj };
    acc[0][j] = b8;
    acc[1][j] = b8;
  }

  for (int k0 = 0; k0 < K; k0 += 32) {
    v16h a0 = load_frag_A(A + (size_t)m0 * K + k0, K, lane);
    v16h a1 = load_frag_A(A + (size_t)(m0 + 16) * K + k0, K, lane);
#pragma unroll
    for (int j = 0; j < 4; ++j) {
      v16h b = load_frag_B(W + (size_t)(n0 + 16 * j) * K + k0, K, lane);
      acc[0][j] = __builtin_amdgcn_wmma_f32_16x16x32_f16(false, a0, false, b, (short)0, acc[0][j], false, false);
      acc[1][j] = __builtin_amdgcn_wmma_f32_16x16x32_f16(false, a1, false, b, (short)0, acc[1][j], false, false);
    }
  }

#pragma unroll
  for (int i = 0; i < 2; ++i)
#pragma unroll
    for (int j = 0; j < 4; ++j) {
      int n  = n0 + 16 * j + (lane & 15);
      int mb = m0 + 16 * i + ((lane >> 4) << 3);
#pragma unroll
      for (int v = 0; v < 8; ++v)
        out[(size_t)(mb + v) * N + n] = acc[i][j][v];
    }
}

// ---------------- persistent recurrent kernel ----------------
// grid = (NWG, 2 cells). Each WG owns HS hidden columns; its Wh slice
// (4 gates x HS rows x 512 K = 256 KB f16) lives in LDS for all 512 steps.
// Staged with CDNA5 GLOBAL_LOAD_ASYNC_TO_LDS_B128 (ASYNCcnt path, no VGPR transit).
// Wave (of 8): mt = wave>>2 (batch half), ht = wave&3 (16-col hidden tile).
extern __shared__ _Float16 lds_wh[];   // 4*HS*H_DIM halves = 256 KB

static __device__ inline void grid_sync(unsigned* bar, unsigned target) {
  __threadfence();
  __syncthreads();
  if (threadIdx.x == 0) {
    __hip_atomic_fetch_add(bar, 1u, __ATOMIC_ACQ_REL, __HIP_MEMORY_SCOPE_AGENT);
    while (__hip_atomic_load(bar, __ATOMIC_ACQUIRE, __HIP_MEMORY_SCOPE_AGENT) < target)
      __builtin_amdgcn_s_sleep(2);
  }
  __syncthreads();
}

__global__ __launch_bounds__(256)
void lstm_rec_kernel(const float* __restrict__ Hx0, const float* __restrict__ Hx1,
                     const _Float16* __restrict__ Wh0, const _Float16* __restrict__ Wh1,
                     const float* __restrict__ mask,
                     _Float16* hbuf0, _Float16* hbuf1,       // 2 parity bufs of [32][512] each
                     unsigned* bar0, unsigned* bar1,
                     _Float16* hist_f16,                     // layer0: out0 [T][B][2H]; else null
                     float* out_f32,                         // layer1: d_out [T][B][2H]; else null
                     int rev0, int rev1) {
  const int cell = blockIdx.y;
  const int wg   = blockIdx.x;
  const int lane = threadIdx.x & 31;
  const int wave = threadIdx.x >> 5;
  const int mt   = wave >> 2;   // 0/1: batch rows [mt*16, mt*16+16)
  const int ht   = wave & 3;    // hidden 16-col tile within this WG's HS slice

  const float*    Hx   = cell ? Hx1  : Hx0;
  const _Float16* Wh   = cell ? Wh1  : Wh0;
  _Float16*       hbuf = cell ? hbuf1 : hbuf0;
  unsigned*       bar  = cell ? bar1 : bar0;
  const int       rev  = cell ? rev1 : rev0;

  // ---- stage this WG's Wh slice into LDS via async DMA (rows: gate g -> g*H + wg*HS .. +HS) ----
  for (int g = 0; g < 4; ++g) {
    const _Float16* s = Wh + ((size_t)g * H_DIM + (size_t)wg * HS) * H_DIM;
    _Float16* d = lds_wh + (size_t)g * HS * H_DIM;
    for (int i = threadIdx.x; i < HS * H_DIM / 8; i += blockDim.x) {
      // VDST = LDS byte offset (low 32 bits of generic LDS pointer), VADDR = 64-bit global addr
      unsigned loff = (unsigned)(uintptr_t)(d + (size_t)i * 8);
      const _Float16* gp = s + (size_t)i * 8;
      asm volatile("global_load_async_to_lds_b128 %0, %1, off"
                   :: "v"(loff), "v"(gp) : "memory");
    }
  }
  asm volatile("s_wait_asynccnt 0x0" ::: "memory");
  __syncthreads();

  const int colb = wg * HS + ht * 16;            // hidden column base (global)
  const int n_h  = colb + (lane & 15);           // this lane's hidden column
  const int mrow = mt * 16 + ((lane >> 4) << 3); // this lane's batch row base (D layout)

  v8f cst = { 0.f, 0.f, 0.f, 0.f, 0.f, 0.f, 0.f, 0.f };

  for (int tt = 0; tt < T_DIM; ++tt) {
    const int tact = rev ? (T_DIM - 1 - tt) : tt;
    const _Float16* hprev = hbuf + (size_t)(tt & 1) * (B_DIM * H_DIM);
    _Float16*       hcur  = hbuf + (size_t)((tt + 1) & 1) * (B_DIM * H_DIM);

    if (tt + 1 < T_DIM) {
      const int tnext = rev ? (T_DIM - 2 - tt) : (tt + 1);
      __builtin_prefetch(Hx + (size_t)tnext * B_DIM * G_DIM + (size_t)mrow * G_DIM, 0, 1);
    }

    // ---- pre-gates: h_prev (32x512) x WhT slice, via WMMA; B from LDS ----
    v8f acc[4];
#pragma unroll
    for (int g = 0; g < 4; ++g) acc[g] = (v8f){ 0.f, 0.f, 0.f, 0.f, 0.f, 0.f, 0.f, 0.f };

    for (int kt = 0; kt < H_DIM / 32; ++kt) {
      v16h a = load_frag_A(hprev + (size_t)(mt * 16) * H_DIM + kt * 32, H_DIM, lane);
#pragma unroll
      for (int g = 0; g < 4; ++g) {
        const _Float16* bp = lds_wh + ((size_t)(g * HS + ht * 16 + (lane & 15))) * H_DIM
                                    + kt * 32 + ((lane >> 4) << 3);
        union { v16h v; half8 h[2]; } ub;
        ub.h[0] = *(const half8*)(bp);
        ub.h[1] = *(const half8*)(bp + 16);
        acc[g] = __builtin_amdgcn_wmma_f32_16x16x32_f16(false, a, false, ub.v, (short)0, acc[g], false, false);
      }
    }

    // ---- add Hx (bias folded in), nonlinearity, state update ----
    const float* hxp = Hx + (size_t)tact * B_DIM * G_DIM + (size_t)mrow * G_DIM;
#pragma unroll
    for (int v = 0; v < 8; ++v) {
      float pi = acc[0][v] + hxp[(size_t)v * G_DIM + 0 * H_DIM + n_h];
      float pf = acc[1][v] + hxp[(size_t)v * G_DIM + 1 * H_DIM + n_h];
      float po = acc[2][v] + hxp[(size_t)v * G_DIM + 2 * H_DIM + n_h];
      float pg = acc[3][v] + hxp[(size_t)v * G_DIM + 3 * H_DIM + n_h];

      float mm = mask[(size_t)tact * B_DIM + mrow + v];
      float hp = (float)hprev[(size_t)(mrow + v) * H_DIM + n_h];

      float cn = cst[v] * sigmoidf_(pf) + tanhf(pg) * sigmoidf_(pi);
      float hn = sigmoidf_(po) * tanhf(cn);       // reference: h uses pre-blend c
      cn = mm * cn + (1.0f - mm) * cst[v];
      hn = mm * hn + (1.0f - mm) * hp;
      cst[v] = cn;

      _Float16 h16 = (_Float16)hn;
      hcur[(size_t)(mrow + v) * H_DIM + n_h] = h16;
      size_t oidx = ((size_t)tact * B_DIM + (mrow + v)) * (2 * H_DIM) + (size_t)cell * H_DIM + n_h;
      if (hist_f16) hist_f16[oidx] = h16;
      if (out_f32)  out_f32[oidx]  = hn;
    }

    grid_sync(bar, (unsigned)(NWG * (tt + 1)));
  }
}

// ---------------- host ----------------
extern "C" void kernel_launch(void* const* d_in, const int* in_sizes, int n_in,
                              void* d_out, int out_size, void* d_ws, size_t ws_size,
                              hipStream_t stream) {
  const float* x     = (const float*)d_in[0];
  const float* mask  = (const float*)d_in[1];
  const float* Wx_f0 = (const float*)d_in[2];
  const float* bx_f0 = (const float*)d_in[3];
  const float* Wh_f0 = (const float*)d_in[4];
  const float* Wx_b0 = (const float*)d_in[5];
  const float* bx_b0 = (const float*)d_in[6];
  const float* Wh_b0 = (const float*)d_in[7];
  const float* Wx_f1 = (const float*)d_in[8];
  const float* bx_f1 = (const float*)d_in[9];
  const float* Wh_f1 = (const float*)d_in[10];
  const float* Wx_b1 = (const float*)d_in[11];
  const float* bx_b1 = (const float*)d_in[12];
  const float* Wh_b1 = (const float*)d_in[13];
  float* out = (float*)d_out;

  char* ws = (char*)d_ws;
  size_t off = 0;
  auto alloc = [&](size_t bytes) -> void* {
    void* p = ws + off;
    off += (bytes + 255) & ~(size_t)255;
    return p;
  };

  unsigned*  bar   = (unsigned*)alloc(256);                                   // 4 counters
  _Float16*  hbuf  = (_Float16*)alloc((size_t)4 * 2 * B_DIM * H_DIM * 2);     // 4 slots x 2 parity
  _Float16*  x16   = (_Float16*)alloc((size_t)T_DIM * B_DIM * D_IN * 2);
  _Float16*  wxf0  = (_Float16*)alloc((size_t)G_DIM * D_IN * 2);
  _Float16*  wxb0  = (_Float16*)alloc((size_t)G_DIM * D_IN * 2);
  _Float16*  whf0  = (_Float16*)alloc((size_t)G_DIM * H_DIM * 2);
  _Float16*  whb0  = (_Float16*)alloc((size_t)G_DIM * H_DIM * 2);
  _Float16*  whf1  = (_Float16*)alloc((size_t)G_DIM * H_DIM * 2);
  _Float16*  whb1  = (_Float16*)alloc((size_t)G_DIM * H_DIM * 2);
  _Float16*  wxf1  = (_Float16*)alloc((size_t)G_DIM * 2 * H_DIM * 2);
  _Float16*  wxb1  = (_Float16*)alloc((size_t)G_DIM * 2 * H_DIM * 2);
  _Float16*  out0  = (_Float16*)alloc((size_t)T_DIM * B_DIM * 2 * H_DIM * 2);
  float*     Hx0   = (float*)alloc((size_t)T_DIM * B_DIM * G_DIM * 4);
  float*     Hx1   = (float*)alloc((size_t)T_DIM * B_DIM * G_DIM * 4);
  (void)ws_size; (void)in_sizes; (void)n_in; (void)out_size;

  (void)hipFuncSetAttribute((const void*)lstm_rec_kernel,
                            hipFuncAttributeMaxDynamicSharedMemorySize, 262144);

  // ---- zero barrier counters + h buffers (deterministic per call) ----
  {
    int nwords = (256 + 4 * 2 * B_DIM * H_DIM * 2) / 4;
    zero32_kernel<<<(nwords + 255) / 256, 256, 0, stream>>>((uint32_t*)ws, nwords);
  }

  // ---- f32 -> f16 conversions ----
  auto conv = [&](const float* s, _Float16* d, size_t n) {
    f32_to_f16_kernel<<<(unsigned)((n + 255) / 256), 256, 0, stream>>>(s, d, (int)n);
  };
  conv(x,     x16,  (size_t)T_DIM * B_DIM * D_IN);
  conv(Wx_f0, wxf0, (size_t)G_DIM * D_IN);
  conv(Wx_b0, wxb0, (size_t)G_DIM * D_IN);
  conv(Wh_f0, whf0, (size_t)G_DIM * H_DIM);
  conv(Wh_b0, whb0, (size_t)G_DIM * H_DIM);
  conv(Wh_f1, whf1, (size_t)G_DIM * H_DIM);
  conv(Wh_b1, whb1, (size_t)G_DIM * H_DIM);
  conv(Wx_f1, wxf1, (size_t)G_DIM * 2 * H_DIM);
  conv(Wx_b1, wxb1, (size_t)G_DIM * 2 * H_DIM);

  const int M = T_DIM * B_DIM;   // 16384

  // ---- layer 0 input projections (bias folded) ----
  gemm_xw_kernel<<<dim3(G_DIM / 128, M / 128), 256, 0, stream>>>(x16, wxf0, bx_f0, Hx0, M, G_DIM, D_IN);
  gemm_xw_kernel<<<dim3(G_DIM / 128, M / 128), 256, 0, stream>>>(x16, wxb0, bx_b0, Hx1, M, G_DIM, D_IN);

  // ---- layer 0 recurrence (fw cell 0, bw cell 1) -> out0 f16 history ----
  lstm_rec_kernel<<<dim3(NWG, 2), 256, 262144, stream>>>(
      Hx0, Hx1, whf0, whb0, mask,
      hbuf + 0 * (2 * B_DIM * H_DIM), hbuf + 1 * (2 * B_DIM * H_DIM),
      bar + 0, bar + 1,
      out0, (float*)nullptr, 0, 1);

  // ---- layer 1 input projections over out0 (K = 2H) ----
  gemm_xw_kernel<<<dim3(G_DIM / 128, M / 128), 256, 0, stream>>>(out0, wxf1, bx_f1, Hx0, M, G_DIM, 2 * H_DIM);
  gemm_xw_kernel<<<dim3(G_DIM / 128, M / 128), 256, 0, stream>>>(out0, wxb1, bx_b1, Hx1, M, G_DIM, 2 * H_DIM);

  // ---- layer 1 recurrence -> d_out fp32 ----
  lstm_rec_kernel<<<dim3(NWG, 2), 256, 262144, stream>>>(
      Hx0, Hx1, whf1, whb1, mask,
      hbuf + 2 * (2 * B_DIM * H_DIM), hbuf + 3 * (2 * B_DIM * H_DIM),
      bar + 2, bar + 3,
      (_Float16*)nullptr, out, 0, 1);
}
